// NTXentLoss_3994319585478
// MI455X (gfx1250) — matrix operand order repaired and checked by reading, not compiled, not run on hardware
//
#include <hip/hip_runtime.h>
#include <math.h>

// ---------------------------------------------------------------------------
// NT-Xent loss, MI455X (gfx1250, wave32).
//   z = concat(z_i, z_j)            [8192, 256] fp32
//   zn = z / max(||z||, 1e-8)
//   sim = zn @ zn^T                 (never materialized; 34 GFLOP)
//   loss = -mean( 2*pos - log( rowsum_{j!=i} exp(2*sim[i,j]) ) )
//
// sim computed with split-bf16 WMMA: x = hi + lo (bf16 each),
//   sim ~= hi.hi + hi.lo + lo.hi   (lo.lo term ~4e-6, dropped)
// Three independent WMMA accumulator chains; B tiles double-buffered in LDS
// via gfx1250 async global->LDS DMA overlapped with the WMMA body.
// ---------------------------------------------------------------------------

typedef __attribute__((ext_vector_type(8)))  __bf16 v8bf;
typedef __attribute__((ext_vector_type(16))) __bf16 v16bf;
typedef __attribute__((ext_vector_type(8)))  float  v8f;
typedef __attribute__((ext_vector_type(4)))  int    v4i;

#define N2      8192    // 2N rows
#define DIMK    256     // feature dim
#define KBLKS   8       // 256 / 32
#define COLCH   8       // column chunks (parallelism over columns)
#define TILES_PER_CHUNK 64   // 8192 / 8 / 16
#define LDS_STRIDE 264  // 256 + 8 pad -> row stride 528B = 132 dwords (4-bank rotate)

#if defined(__gfx1250__) && __has_builtin(__builtin_amdgcn_global_load_async_to_lds_b128)
#define HAVE_ASYNC_LDS 1
// src: global (AS1) v4i*, dst: LDS (AS3) v4i*, imm offset, imm cpol
#define ASYNC_CP16(dst, src)                                                   \
  __builtin_amdgcn_global_load_async_to_lds_b128(                              \
      (__attribute__((address_space(1))) v4i*)(src),                           \
      (__attribute__((address_space(3))) v4i*)(dst), 0, 0)
#endif

// ---------------------------------------------------------------------------
// Kernel 1: row normalize, split into bf16 hi/lo, zero rowsum accumulators.
// One wave per row (32 lanes x 8 floats).
// ---------------------------------------------------------------------------
__global__ __launch_bounds__(256) void k_prep(const float* __restrict__ zi,
                                              const float* __restrict__ zj,
                                              __bf16* __restrict__ hi,
                                              __bf16* __restrict__ lo,
                                              float* __restrict__ rowsum) {
  const int lane = threadIdx.x & 31;
  const int wave = threadIdx.x >> 5;
  const int row  = blockIdx.x * 8 + wave;

  const float* src = (row < (N2 / 2)) ? (zi + (size_t)row * DIMK)
                                      : (zj + (size_t)(row - N2 / 2) * DIMK);
  float4 x0 = ((const float4*)src)[lane * 2 + 0];
  float4 x1 = ((const float4*)src)[lane * 2 + 1];
  float v[8] = {x0.x, x0.y, x0.z, x0.w, x1.x, x1.y, x1.z, x1.w};

  float ss = 0.f;
#pragma unroll
  for (int i = 0; i < 8; ++i) ss += v[i] * v[i];
#pragma unroll
  for (int m = 16; m >= 1; m >>= 1) ss += __shfl_xor(ss, m, 32);

  const float inv = 1.0f / fmaxf(sqrtf(ss), 1e-8f);

  v8bf hv{}, lv{};
#pragma unroll
  for (int i = 0; i < 8; ++i) {
    float zn = v[i] * inv;
    __bf16 h = (__bf16)zn;
    hv[i] = h;
    lv[i] = (__bf16)(zn - (float)h);
  }
  *(v8bf*)(hi + (size_t)row * DIMK + lane * 8) = hv;
  *(v8bf*)(lo + (size_t)row * DIMK + lane * 8) = lv;
  if (lane == 0) rowsum[row] = 0.0f;
}

// ---------------------------------------------------------------------------
// Kernel 2: pos[i] = <zn[i], zn[(i+N)%2N]>, self[i] = <zn[i], zn[i]>.
// zn reconstructed as hi+lo (fp32-accurate).
// ---------------------------------------------------------------------------
__global__ __launch_bounds__(256) void k_pos(const __bf16* __restrict__ hi,
                                             const __bf16* __restrict__ lo,
                                             float* __restrict__ pos,
                                             float* __restrict__ selfd) {
  const int lane = threadIdx.x & 31;
  const int wave = threadIdx.x >> 5;
  const int row  = blockIdx.x * 8 + wave;
  const int par  = (row + N2 / 2) & (N2 - 1);

  v8bf ah = *(const v8bf*)(hi + (size_t)row * DIMK + lane * 8);
  v8bf al = *(const v8bf*)(lo + (size_t)row * DIMK + lane * 8);
  v8bf bh = *(const v8bf*)(hi + (size_t)par * DIMK + lane * 8);
  v8bf bl = *(const v8bf*)(lo + (size_t)par * DIMK + lane * 8);

  float p = 0.f, s = 0.f;
#pragma unroll
  for (int i = 0; i < 8; ++i) {
    float a = (float)ah[i] + (float)al[i];
    float b = (float)bh[i] + (float)bl[i];
    p += a * b;
    s += a * a;
  }
#pragma unroll
  for (int m = 16; m >= 1; m >>= 1) {
    p += __shfl_xor(p, m, 32);
    s += __shfl_xor(s, m, 32);
  }
  if (lane == 0) { pos[row] = p; selfd[row] = s; }
}

// ---------------------------------------------------------------------------
// Kernel 3: fused Gram + exp row-sums via WMMA.
// Block = 8 waves; wave w owns row-strip (blockIdx.x&63)*8 + w (16 rows).
// blockIdx.x>>6 selects one of 8 column chunks (1024 cols = 64 tiles).
// B tiles (hi+lo, full K) double-buffered in LDS, shared by all 8 waves;
// async DMA for tile t+1 overlaps the WMMA body of tile t.
// ---------------------------------------------------------------------------
__global__ __launch_bounds__(256) void k_main(const __bf16* __restrict__ hi,
                                              const __bf16* __restrict__ lo,
                                              float* __restrict__ rowsum) {
  __shared__ __align__(16) __bf16 sBhi[2][16][LDS_STRIDE];
  __shared__ __align__(16) __bf16 sBlo[2][16][LDS_STRIDE];

  const int lane     = threadIdx.x & 31;
  const int wave     = threadIdx.x >> 5;
  const int rsGroup  = blockIdx.x & 63;
  const int colChunk = blockIdx.x >> 6;
  const int i0       = (rsGroup * 8 + wave) * 16;
  const int jbase    = colChunk * TILES_PER_CHUNK * 16;

  // staging pattern: thread t copies 16 bf16 of row t/16 at K = (t%16)*16
  const int stR = threadIdx.x >> 4;
  const int stS = (threadIdx.x & 15) * 16;

  // issue the async DMA for buffer `buf` <- column tile `ct`
  auto stage = [&](int buf, int ct) {
    const int j0 = jbase + ct * 16;
    const __bf16* gh = hi + (size_t)(j0 + stR) * DIMK + stS;
    const __bf16* gl = lo + (size_t)(j0 + stR) * DIMK + stS;
    __bf16* dh = &sBhi[buf][stR][stS];
    __bf16* dl = &sBlo[buf][stR][stS];
#ifdef HAVE_ASYNC_LDS
    ASYNC_CP16(dh,     gh);
    ASYNC_CP16(dh + 8, gh + 8);
    ASYNC_CP16(dl,     gl);
    ASYNC_CP16(dl + 8, gl + 8);
#else
    *(v8bf*)dh       = *(const v8bf*)gh;
    *(v8bf*)(dh + 8) = *(const v8bf*)(gh + 8);
    *(v8bf*)dl       = *(const v8bf*)gl;
    *(v8bf*)(dl + 8) = *(const v8bf*)(gl + 8);
#endif
  };

  // wait for this wave's outstanding DMAs, then block-wide barrier: the
  // barrier's workgroup fence also guarantees the previous tile's LDS
  // readers (dscnt) are drained before the buffer is overwritten.
  auto wait_sync = [&]() {
#ifdef HAVE_ASYNC_LDS
    asm volatile("s_wait_asynccnt 0" ::: "memory");
#endif
    __syncthreads();
  };

  // kick off tile 0's DMA before loading A so it overlaps the A loads
  stage(0, 0);

  // ---- A fragments (this wave's 16 rows, all K), kept in VGPRs ----
  // ISA A 16x32 bf16 layout: lanes 0-15 = row M, K 0-7 & 16-23;
  //                          lanes 16-31 = row M, K 8-15 & 24-31.
  const __bf16* hrow = hi + (size_t)(i0 + (lane & 15)) * DIMK;
  const __bf16* lrow = lo + (size_t)(i0 + (lane & 15)) * DIMK;
  const int aoff = (lane < 16) ? 0 : 8;

  v16bf Ahi[KBLKS], Alo[KBLKS];
#pragma unroll
  for (int kb = 0; kb < KBLKS; ++kb) {
    v8bf h0 = *(const v8bf*)(hrow + kb * 32 + aoff);
    v8bf h1 = *(const v8bf*)(hrow + kb * 32 + 16 + aoff);
    Ahi[kb] = __builtin_shufflevector(h0, h1, 0,1,2,3,4,5,6,7,8,9,10,11,12,13,14,15);
    v8bf l0 = *(const v8bf*)(lrow + kb * 32 + aoff);
    v8bf l1 = *(const v8bf*)(lrow + kb * 32 + 16 + aoff);
    Alo[kb] = __builtin_shufflevector(l0, l1, 0,1,2,3,4,5,6,7,8,9,10,11,12,13,14,15);
  }

  float accs[8] = {0.f, 0.f, 0.f, 0.f, 0.f, 0.f, 0.f, 0.f};

  // B read pattern: lane l -> column n = l%16 (= zn row j0+n);
  //                 lanes 0-15 K 0-7 & 16-23, lanes 16-31 K 8-15 & 24-31 (per k-block).
  const int brow = lane & 15;
  const int boff = (lane < 16) ? 0 : 8;

  // consume buffer `buf`: 3 independent WMMA chains (hi.hi, hi.lo, lo.hi)
  auto consume = [&](int buf) {
    v8f C0 = {0.f, 0.f, 0.f, 0.f, 0.f, 0.f, 0.f, 0.f};
    v8f C1 = {0.f, 0.f, 0.f, 0.f, 0.f, 0.f, 0.f, 0.f};
    v8f C2 = {0.f, 0.f, 0.f, 0.f, 0.f, 0.f, 0.f, 0.f};
#pragma unroll
    for (int kb = 0; kb < KBLKS; ++kb) {
      const __bf16* bh = &sBhi[buf][brow][kb * 32 + boff];
      const __bf16* bl = &sBlo[buf][brow][kb * 32 + boff];
      v8bf b0 = *(const v8bf*)bh;
      v8bf b1 = *(const v8bf*)(bh + 16);
      v16bf Bhi = __builtin_shufflevector(b0, b1, 0,1,2,3,4,5,6,7,8,9,10,11,12,13,14,15);
      b0 = *(const v8bf*)bl;
      b1 = *(const v8bf*)(bl + 16);
      v16bf Blo = __builtin_shufflevector(b0, b1, 0,1,2,3,4,5,6,7,8,9,10,11,12,13,14,15);

      C0 = __builtin_amdgcn_wmma_f32_16x16x32_bf16(false, Ahi[kb], false, Bhi,
                                                   (short)0, C0, false, false);
      C1 = __builtin_amdgcn_wmma_f32_16x16x32_bf16(false, Ahi[kb], false, Blo,
                                                   (short)0, C1, false, false);
      C2 = __builtin_amdgcn_wmma_f32_16x16x32_bf16(false, Alo[kb], false, Bhi,
                                                   (short)0, C2, false, false);
    }
    // fuse exp(sim/T), T=0.5 -> exp(2x); accumulate per-row partial sums
#pragma unroll
    for (int r = 0; r < 8; ++r)
      accs[r] += __expf(2.0f * (C0[r] + C1[r] + C2[r]));
  };

  // double-buffered main loop (manually unrolled x2 so buffer ids are static)
  for (int ct = 0; ct < TILES_PER_CHUNK; ct += 2) {
    wait_sync();                                     // buf0[ct] ready
    if (ct + 1 < TILES_PER_CHUNK) stage(1, ct + 1);  // DMA next into buf1
    consume(0);
    wait_sync();                                     // buf1[ct+1] ready
    if (ct + 2 < TILES_PER_CHUNK) stage(0, ct + 2);  // DMA next into buf0
    consume(1);
  }

  // D layout: VGPR r, lanes 0-15 -> row i0+r; lanes 16-31 -> row i0+8+r.
  // Reduce the 16 column-lanes of each half, then atomically merge chunks.
#pragma unroll
  for (int r = 0; r < 8; ++r) {
    float s = accs[r];
    s += __shfl_xor(s, 1, 16);
    s += __shfl_xor(s, 2, 16);
    s += __shfl_xor(s, 4, 16);
    s += __shfl_xor(s, 8, 16);
    if (lane == 0)  atomicAdd(&rowsum[i0 + r], s);
    if (lane == 16) atomicAdd(&rowsum[i0 + 8 + r], s);
  }
}

// ---------------------------------------------------------------------------
// Kernel 4: loss = -mean( 2*pos[i] - log( rowsum[i] - exp(2*self[i]) ) )
// ---------------------------------------------------------------------------
__global__ __launch_bounds__(256) void k_final(const float* __restrict__ rowsum,
                                               const float* __restrict__ pos,
                                               const float* __restrict__ selfd,
                                               float* __restrict__ out) {
  __shared__ float red[256];
  float t = 0.f;
  for (int i = threadIdx.x; i < N2; i += 256) {
    float denom = rowsum[i] - __expf(2.0f * selfd[i]);  // drop j==i term
    t += 2.0f * pos[i] - logf(denom);
  }
  red[threadIdx.x] = t;
  __syncthreads();
  for (int s = 128; s > 0; s >>= 1) {
    if (threadIdx.x < s) red[threadIdx.x] += red[threadIdx.x + s];
    __syncthreads();
  }
  if (threadIdx.x == 0) out[0] = -red[0] / (float)N2;
}

// ---------------------------------------------------------------------------
extern "C" void kernel_launch(void* const* d_in, const int* in_sizes, int n_in,
                              void* d_out, int out_size, void* d_ws, size_t ws_size,
                              hipStream_t stream) {
  const float* zi = (const float*)d_in[0];
  const float* zj = (const float*)d_in[1];

  char* ws = (char*)d_ws;
  __bf16* hi     = (__bf16*)(ws);                                  // 4 MB
  __bf16* lo     = (__bf16*)(ws + (size_t)(4 << 20));              // 4 MB
  float*  rowsum = (float*)(ws + (size_t)(8 << 20));               // 32 KB
  float*  pos    = (float*)(ws + (size_t)(8 << 20) + (32 << 10));  // 32 KB
  float*  selfd  = (float*)(ws + (size_t)(8 << 20) + (64 << 10));  // 32 KB
  float*  out    = (float*)d_out;

  k_prep <<<N2 / 8, 256, 0, stream>>>(zi, zj, hi, lo, rowsum);
  k_pos  <<<N2 / 8, 256, 0, stream>>>(hi, lo, pos, selfd);
  k_main <<<(N2 / 16 / 8) * COLCH, 256, 0, stream>>>(hi, lo, rowsum);
  k_final<<<1, 256, 0, stream>>>(rowsum, pos, selfd, out);
}
